// UserEncoder_40424232190525
// MI455X (gfx1250) — matrix operand order repaired and checked by reading, not compiled, not run on hardware
//
#include <hip/hip_runtime.h>
#include <hip/hip_bf16.h>
#include <stdint.h>

// ---- sizes from the reference ----
#define B_    64
#define HIS_  50
#define S_    32
#define E_    768
#define C_    18
#define CLICK_ 100

typedef __attribute__((ext_vector_type(16))) _Float16 v16h;
typedef __attribute__((ext_vector_type(8)))  float    v8f;

// -------------------------------------------------------------------------
// K1: embedding gather + masked mean pool -> f16 pooled [B*H, E]
// one block per (b,h) row; 192 threads x float4 (bandwidth-critical kernel:
// every load is a full global_load_b128)
// -------------------------------------------------------------------------
__global__ __launch_bounds__(192) void pool_kernel(
    const int* __restrict__ enc, const unsigned char* __restrict__ amask,
    const float* __restrict__ wemb, _Float16* __restrict__ pooledH)
{
    const int bh  = blockIdx.x;                // 0..3199
    const int tid = threadIdx.x;               // 0..191 -> cols 4*tid..4*tid+3
    const int* e           = enc   + (size_t)bh * S_;
    const unsigned char* m = amask + (size_t)bh * S_;

    float4 acc = make_float4(0.f, 0.f, 0.f, 0.f);
    float  cnt = 0.f;
    for (int s = 0; s < S_; ++s) {
        const float mv = m[s] ? 1.f : 0.f;
        cnt += mv;
        if (mv != 0.f) {
            const float4 v = *(const float4*)(wemb + (size_t)e[s] * E_ + 4 * tid);
            acc.x += v.x; acc.y += v.y; acc.z += v.z; acc.w += v.w;
        }
    }
    const float inv = 1.f / fmaxf(cnt, 1.f);
    union { _Float16 h[4]; uint2 u; } o;
    o.h[0] = (_Float16)(acc.x * inv);
    o.h[1] = (_Float16)(acc.y * inv);
    o.h[2] = (_Float16)(acc.z * inv);
    o.h[3] = (_Float16)(acc.w * inv);
    *(uint2*)(pooledH + (size_t)bh * E_ + 4 * tid) = o.u;   // 8B store
}

// -------------------------------------------------------------------------
// K1b: proj_w [K][N] f32 -> Bt [N][K] f16 (K-contiguous for B-operand loads)
// -------------------------------------------------------------------------
__global__ __launch_bounds__(256) void wtrans_kernel(
    const float* __restrict__ W, _Float16* __restrict__ Bt)
{
    const int n = blockIdx.x;                  // 0..767
    #pragma unroll
    for (int i = 0; i < 3; ++i) {
        const int k = threadIdx.x + 256 * i;
        Bt[(size_t)n * E_ + k] = (_Float16)W[(size_t)k * E_ + n];
    }
}

// -------------------------------------------------------------------------
// K2: news_repr = tanh(pooled @ proj_w + b) via v_wmma_f32_16x16x32_f16
// One wave per 16x64 output block (4 accumulators, A reused 4x per k-step).
// Grid (200, 3), block 128 (4 waves). Per-lane packing follows CDNA5 ISA
// 7.12.2 16-bit A / B layouts.
// -------------------------------------------------------------------------
__global__ __launch_bounds__(128) void gemm_tanh_kernel(
    const _Float16* __restrict__ A,   // [3200, 768] row-major f16
    const _Float16* __restrict__ Bt,  // [768(N), 768(K)] f16
    const float* __restrict__ bias,   // [768]
    float* __restrict__ Cout)         // [3200, 768] f32
{
    const int wave  = threadIdx.x >> 5;
    const int lane  = threadIdx.x & 31;
    const int mTile = blockIdx.x;                       // 0..199
    const int nBase = (blockIdx.y * 4 + wave) * 64;     // 0..704 step 64
    const int hi    = lane >> 4;                        // lane half
    const int l15   = lane & 15;
    const int mRow  = mTile * 16 + l15;                 // A row (M = lane%16)

    const _Float16* arow = A + (size_t)mRow * E_;
    const _Float16* brow[4];
    #pragma unroll
    for (int j = 0; j < 4; ++j)
        brow[j] = Bt + (size_t)(nBase + j * 16 + l15) * E_;

    v8f c[4];
    #pragma unroll
    for (int j = 0; j < 4; ++j) c[j] = (v8f){0.f,0.f,0.f,0.f,0.f,0.f,0.f,0.f};

    for (int k0 = 0; k0 < E_; k0 += 32) {
        union { v16h v; uint32_t u[8]; } a;
        #pragma unroll
        for (int v = 0; v < 8; ++v) {
            // A 16x32 f16 layout: lanes0-15: V0..3 -> K=0..7, V4..7 -> K=16..23
            //                     lanes16-31: +8 (low VGPRs) / +8 (high VGPRs)
            const int kA = k0 + ((v < 4) ? ((hi ? 8 : 0) + 2 * v)
                                         : ((hi ? 24 : 16) + 2 * (v - 4)));
            a.u[v] = *(const uint32_t*)(arow + kA);     // [15:0]=K, [31:16]=K+1
        }
        #pragma unroll
        for (int j = 0; j < 4; ++j) {
            union { v16h v; uint32_t u[8]; } b;
            #pragma unroll
            for (int v = 0; v < 8; ++v) {
                // B 32x16 f16 layout: same K mapping as A (V0..3 -> K=0..7, V4..7 -> K=16..23, +8 hi)
                const int kB = k0 + ((v < 4) ? ((hi ? 8 : 0) + 2 * v)
                                             : ((hi ? 24 : 16) + 2 * (v - 4)));
                b.u[v] = *(const uint32_t*)(brow[j] + kB);
            }
            c[j] = __builtin_amdgcn_wmma_f32_16x16x32_f16(
                       /*neg_a=*/false, a.v, /*neg_b=*/false, b.v,
                       /*c_mod=*/(short)0, c[j], /*reuse_a=*/false, /*reuse_b=*/false);
        }
    }

    #pragma unroll
    for (int j = 0; j < 4; ++j) {
        const int n  = nBase + j * 16 + l15;            // N = lane%16
        const float bn = bias[n];
        #pragma unroll
        for (int r = 0; r < 8; ++r) {
            // 32-bit C/D layout: VGPR r -> M = r (lanes0-15) or r+8 (lanes16-31)
            const int mm = mTile * 16 + r + (hi ? 8 : 0);
            Cout[(size_t)mm * E_ + n] = tanhf(c[j][r] + bn);
        }
    }
}

// -------------------------------------------------------------------------
// K3: news_w[row] = dot(news_repr[row], w_news_attn); one block per row
// -------------------------------------------------------------------------
__global__ __launch_bounds__(256) void newsw_kernel(
    const float* __restrict__ news, const float* __restrict__ wv,
    float* __restrict__ out)
{
    __shared__ float red[256];
    const int row = blockIdx.x;
    const int tid = threadIdx.x;
    float s = 0.f;
    #pragma unroll
    for (int i = 0; i < 3; ++i) {
        const int e = tid + 256 * i;
        s += news[(size_t)row * E_ + e] * wv[e];
    }
    red[tid] = s; __syncthreads();
    for (int off = 128; off > 0; off >>= 1) {
        if (tid < off) red[tid] += red[tid + off];
        __syncthreads();
    }
    if (tid == 0) out[row] = red[0];
}

// -------------------------------------------------------------------------
// K4: per-(b,c) masked softmax over history + weighted combine
//     -> category_repr (to d_out) + score[b,c] (cat_w + click, -inf if empty)
// -------------------------------------------------------------------------
__global__ __launch_bounds__(256) void cat_kernel(
    const unsigned char* __restrict__ cmask,   // [B, C, H]
    const float* __restrict__ newsW,           // [B*H]
    const float* __restrict__ news,            // [B*H, E]
    const float* __restrict__ catEmb,          // [C, E]
    const float* __restrict__ wcat,            // [E]
    const float* __restrict__ clickEmb,        // [H+1, 100]
    const float* __restrict__ clickW,          // [100]
    const float* __restrict__ clickB,          // [1]
    float* __restrict__ catRepr,               // d_out region [B*C, E]
    float* __restrict__ scores)                // [B*C]
{
    const int bc  = blockIdx.x;
    const int b   = bc / C_;
    const int cI  = bc % C_;
    const int tid = threadIdx.x;

    __shared__ float attn[HIS_ + 1];
    __shared__ float red[256];
    __shared__ int   scount;

    if (tid == 0) {
        const unsigned char* m = cmask + (size_t)bc * HIS_;
        float mx = -INFINITY; int cnt = 0;
        for (int h = 0; h < HIS_; ++h)
            if (m[h]) { cnt++; mx = fmaxf(mx, newsW[b * HIS_ + h]); }
        float sum = 0.f;
        for (int h = 0; h < HIS_; ++h) {
            const float w = m[h] ? expf(newsW[b * HIS_ + h] - mx) : 0.f;
            attn[h] = w; sum += w;
        }
        const float inv = (cnt > 0) ? 1.f / sum : 0.f;   // empty cat -> attn 0
        for (int h = 0; h < HIS_; ++h) attn[h] *= inv;
        scount = cnt;
    }
    __syncthreads();

    float acc[3] = {0.f, 0.f, 0.f};
    for (int h = 0; h < HIS_; ++h) {
        const float w = attn[h];
        if (w != 0.f) {
            const float* row = news + (size_t)(b * HIS_ + h) * E_;
            acc[0] += w * row[tid];
            acc[1] += w * row[tid + 256];
            acc[2] += w * row[tid + 512];
        }
    }

    float dotp = 0.f;
    #pragma unroll
    for (int i = 0; i < 3; ++i) {
        const int e = tid + 256 * i;
        const float v = acc[i] + catEmb[(size_t)cI * E_ + e];
        catRepr[(size_t)bc * E_ + e] = v;
        dotp += v * wcat[e];
    }
    red[tid] = dotp; __syncthreads();
    for (int off = 128; off > 0; off >>= 1) {
        if (tid < off) red[tid] += red[tid + off];
        __syncthreads();
    }
    if (tid == 0) {
        const int cnt = scount;
        float click = clickB[0];
        const float* ce = clickEmb + (size_t)cnt * CLICK_;
        for (int j = 0; j < CLICK_; ++j) click += ce[j] * clickW[j];
        scores[bc] = (cnt == 0) ? -INFINITY : (red[0] + click);
    }
}

// -------------------------------------------------------------------------
// K5: per-b softmax over categories + user_repr = final @ category_repr
// -------------------------------------------------------------------------
__global__ __launch_bounds__(256) void user_kernel(
    const float* __restrict__ scores, const float* __restrict__ catRepr,
    float* __restrict__ userOut)
{
    const int b   = blockIdx.x;
    const int tid = threadIdx.x;

    float w[C_];
    float mx = -INFINITY;
    for (int c = 0; c < C_; ++c) mx = fmaxf(mx, scores[b * C_ + c]);
    float sum = 0.f;
    for (int c = 0; c < C_; ++c) {
        const float v = (mx == -INFINITY) ? 0.f : expf(scores[b * C_ + c] - mx);
        w[c] = v; sum += v;
    }
    const float inv = (sum > 0.f) ? 1.f / sum : 0.f;

    float acc[3] = {0.f, 0.f, 0.f};
    for (int c = 0; c < C_; ++c) {
        const float ww = w[c] * inv;
        const float* row = catRepr + (size_t)(b * C_ + c) * E_;
        acc[0] += ww * row[tid];
        acc[1] += ww * row[tid + 256];
        acc[2] += ww * row[tid + 512];
    }
    userOut[(size_t)b * E_ + tid]       = acc[0];
    userOut[(size_t)b * E_ + tid + 256] = acc[1];
    userOut[(size_t)b * E_ + tid + 512] = acc[2];
}

// -------------------------------------------------------------------------
extern "C" void kernel_launch(void* const* d_in, const int* in_sizes, int n_in,
                              void* d_out, int out_size, void* d_ws, size_t ws_size,
                              hipStream_t stream)
{
    (void)in_sizes; (void)n_in; (void)out_size; (void)ws_size;

    const int*           enc      = (const int*)d_in[0];
    const unsigned char* amask    = (const unsigned char*)d_in[1];
    const unsigned char* cmask    = (const unsigned char*)d_in[2];
    const float*         wemb     = (const float*)d_in[3];
    const float*         projW    = (const float*)d_in[4];
    const float*         projB    = (const float*)d_in[5];
    const float*         catEmb   = (const float*)d_in[6];
    const float*         wNews    = (const float*)d_in[7];
    const float*         wCat     = (const float*)d_in[8];
    const float*         clickEmb = (const float*)d_in[9];
    const float*         clickW   = (const float*)d_in[10];
    const float*         clickB   = (const float*)d_in[11];

    const int rows = B_ * HIS_;    // 3200

    // workspace layout (byte offsets, 16B aligned)
    char* ws = (char*)d_ws;
    _Float16* pooledH = (_Float16*)(ws + 0);                         // 3200*768*2 = 4,915,200
    _Float16* Bt      = (_Float16*)(ws + 4915200);                   // 768*768*2  = 1,179,648
    float*    newsR   = (float*)   (ws + 6094848);                   // 3200*768*4 = 9,830,400
    float*    newsW_  = (float*)   (ws + 15925248);                  // 3200*4
    float*    scores  = (float*)   (ws + 15938048);                  // 1152*4

    float* catRepr = (float*)d_out;                                  // [64*18, 768]
    float* userOut = (float*)d_out + (size_t)B_ * C_ * E_;           // [64, 768]

    pool_kernel  <<<rows, 192, 0, stream>>>(enc, amask, wemb, pooledH);
    wtrans_kernel<<<E_,   256, 0, stream>>>(projW, Bt);

    dim3 ggrid(rows / 16, 3);   // 200 M-tiles x (3 blocks * 4 waves * 64 cols = 768)
    gemm_tanh_kernel<<<ggrid, 128, 0, stream>>>(pooledH, Bt, projB, newsR);

    newsw_kernel<<<rows,    256, 0, stream>>>(newsR, wNews, newsW_);
    cat_kernel  <<<B_ * C_, 256, 0, stream>>>(cmask, newsW_, newsR, catEmb, wCat,
                                              clickEmb, clickW, clickB,
                                              catRepr, scores);
    user_kernel <<<B_,      256, 0, stream>>>(scores, catRepr, userOut);
}
